// GroupViTCrossAttentionLayer_12610023981541
// MI455X (gfx1250) — compile-verified, matrix operand order checked by reading, not run on hardware
//
#include <hip/hip_runtime.h>
#include <hip/hip_bf16.h>
#include <math.h>

typedef __attribute__((ext_vector_type(16))) _Float16 v16h;
typedef __attribute__((ext_vector_type(8)))  _Float16 v8h;
typedef __attribute__((ext_vector_type(8)))  float    v8f;
typedef unsigned int u32;
typedef __attribute__((ext_vector_type(4))) u32 u32x4;
typedef __attribute__((ext_vector_type(8))) int i32x8;
typedef __attribute__((ext_vector_type(4))) int i32x4;

namespace {
constexpr int kB = 16, kT = 512, kS = 2048, kD = 768, kH = 12, kFF = 3072;
constexpr float kScale = 0.125f; // (64)^-0.5
constexpr int kKP = 128;         // k-panel staged in LDS per step
constexpr int kPanelHalves = 64 * kKP;           // 8192 f16
constexpr u32 kPanelBytes  = kPanelHalves * 2;   // 16 KB
}

// ---------------------------------------------------------------------------
// WMMA helpers
// ---------------------------------------------------------------------------
__device__ __forceinline__ v8f wmma_f16(v16h a, v16h b, v8f c) {
  return __builtin_amdgcn_wmma_f32_16x16x32_f16(false, a, false, b, (short)0, c,
                                                false, false);
}

// 16x32 f16 tile (row-major, leading dim `ld` elements) -> A-operand layout
// (identical register image serves as B operand of the tile's transpose).
__device__ __forceinline__ v16h load_tile16x32(const _Float16* __restrict__ p0, int ld) {
  const int lane = threadIdx.x & 31;
  const int row  = lane & 15;
  const int kb   = (lane >> 4) << 3;
  const _Float16* p = p0 + (size_t)row * ld + kb;
  v8h lo = *(const v8h*)(p);
  v8h hi = *(const v8h*)(p + 16);
  v16h r;
#pragma unroll
  for (int i = 0; i < 8; ++i) { r[i] = lo[i]; r[i + 8] = hi[i]; }
  return r;
}

// ---------------------------------------------------------------------------
// Tensor Data Mover staging of a 64-row x kKP-col f16 panel into LDS.
// Descriptor per CDNA5 ISA ch.7/8 (D# groups 0..3). One wave issues; the
// whole block consumes after s_wait_tensorcnt + barrier.
// This toolchain's builtin is the 6-arg clang-23 form:
//   (u32x4 g0, i32x8 g1, i32x4 g2, i32x4 g3, i32x8 pad, i32 cpol)
// ---------------------------------------------------------------------------
__device__ __forceinline__ void stage_panel(const _Float16* __restrict__ gsrc,
                                            _Float16* __restrict__ lpanel,
                                            u32 lds_byte_off, int ld) {
#if __has_builtin(__builtin_amdgcn_tensor_load_to_lds)
  (void)lpanel;
  if ((threadIdx.x >> 5) == 0) {
    const unsigned long long ga = (unsigned long long)(uintptr_t)gsrc;
    u32x4 g0;
    g0[0] = 1u;                                 // count=1 (valid user D#)
    g0[1] = lds_byte_off;                       // lds_addr (group segment bytes)
    g0[2] = (u32)(ga & 0xffffffffu);            // global_addr[31:0]
    g0[3] = (u32)((ga >> 32) & 0x1ffffffu)      // global_addr[56:32]
            | (2u << 30);                       // type = 2 ("image")
    const u32 dim0 = (u32)ld;                   // tensor dim0 = K (elements)
    const u32 dim1 = 64u;                       // rows
    const u32 str0 = (u32)ld;                   // row stride (elements)
    i32x8 g1;
    g1[0] = (int)(1u << 16);                    // workgroup_mask=0, data_size=2B
    g1[1] = (int)((dim0 & 0xffffu) << 16);      // tensor_dim0[15:0]
    g1[2] = (int)(((dim0 >> 16) & 0xffffu) | ((dim1 & 0xffffu) << 16));
    g1[3] = (int)(((dim1 >> 16) & 0xffffu) | (((u32)kKP & 0xffffu) << 16)); // tile_dim0
    g1[4] = (int)(64u);                         // tile_dim1=64, tile_dim2=0
    g1[5] = (int)str0;                          // tensor_dim0_stride[31:0]
    g1[6] = 0;                                  // stride[47:32], dim1_stride lo
    g1[7] = 0;
    i32x4 z4 = {};
    i32x8 z8 = {};
    __builtin_amdgcn_tensor_load_to_lds(g0, g1, z4, z4, z8, 0);
  }
#else
  (void)lds_byte_off;
  for (int idx = threadIdx.x; idx < kPanelHalves / 8; idx += 256) {
    const int row = idx >> 4;
    const int c   = (idx & 15) << 3;
    *(v8h*)(lpanel + row * kKP + c) = *(const v8h*)(gsrc + (size_t)row * ld + c);
  }
#endif
}

__device__ __forceinline__ void stage_wait() {
#if __has_builtin(__builtin_amdgcn_tensor_load_to_lds)
  if ((threadIdx.x >> 5) == 0) {
#if __has_builtin(__builtin_amdgcn_s_wait_tensorcnt)
    __builtin_amdgcn_s_wait_tensorcnt(0);
#else
    asm volatile("s_wait_tensorcnt 0x0" ::: "memory");
#endif
  }
#endif
}

// ---------------------------------------------------------------------------
// f32 -> f16 conversion
// ---------------------------------------------------------------------------
__global__ void cvt_f32_f16(const float* __restrict__ in, _Float16* __restrict__ out,
                            size_t n) {
  size_t i = (size_t)blockIdx.x * blockDim.x + threadIdx.x;
  if (i < n) out[i] = (_Float16)in[i];
}

// ---------------------------------------------------------------------------
// WMMA GEMM with TDM-staged weight panels.
// Block (8 waves) computes a 128x64 tile: each wave one 16x64 strip; the
// 64xK weight strip is consumed as double-buffered 64x128 LDS panels.
//   out = epilogue(A[M,K] @ W[N,K]^T + bias)
// Grids are exact: M%128==0, N%64==0, K%128==0 for every call site.
// ---------------------------------------------------------------------------
enum { EPI_F16 = 0, EPI_F16_SCALE = 1, EPI_F16_TRANSV = 2, EPI_F32_RESID = 3,
       EPI_F16_GELU = 4 };

template <int MODE>
__global__ __launch_bounds__(256) void gemm_wmma(
    const _Float16* __restrict__ A,   // [M,K] f16 row-major
    const _Float16* __restrict__ W,   // [N,K] f16 row-major
    const float* __restrict__ bias,   // [N] f32
    void* __restrict__ out,
    const float* __restrict__ resid,  // [M,N] f32 (EPI_F32_RESID)
    int M, int N, int K, float scale, int S) {
  __shared__ _Float16 panel[2][kPanelHalves];   // 2 x 16KB (group offset 0)
  const int blocks_n = N >> 6;
  const int bm = blockIdx.x / blocks_n;
  const int bn = blockIdx.x % blocks_n;
  const int wib = threadIdx.x >> 5;
  const int lane = threadIdx.x & 31;
  const int hi = lane >> 4;
  const int nn = lane & 15;

  const int tmrow = bm * 128 + wib * 16;          // this wave's 16 rows
  const _Float16* a0 = A + (size_t)tmrow * K;
  const _Float16* wbase = W + (size_t)bn * 64 * K;

  stage_panel(wbase, panel[0], 0, K);

  v8f acc[4] = {{}, {}, {}, {}};
  const int npan = K / kKP;
  for (int ip = 0; ip < npan; ++ip) {
    stage_wait();
    __syncthreads();                               // publish panel[ip&1]
    if (ip + 1 < npan)                             // prefetch next (buffer free:
      stage_panel(wbase + (size_t)(ip + 1) * kKP,  //  consumed before last barrier)
                  panel[(ip + 1) & 1], ((ip + 1) & 1) ? kPanelBytes : 0, K);
    const _Float16* pn = panel[ip & 1];
    const _Float16* ak = a0 + ip * kKP;
#pragma unroll
    for (int kk = 0; kk < kKP; kk += 32) {
      const v16h av = load_tile16x32(ak + kk, K);
#pragma unroll
      for (int j = 0; j < 4; ++j) {
        const v16h bv = load_tile16x32(pn + j * 16 * kKP + kk, kKP);
        acc[j] = wmma_f16(av, bv, acc[j]);
      }
    }
    __syncthreads();                               // done with panel[ip&1]
  }

#pragma unroll
  for (int j = 0; j < 4; ++j) {
    const int n = bn * 64 + j * 16 + nn;
    const float bval = bias ? bias[n] : 0.0f;
    if constexpr (MODE == EPI_F16_TRANSV) {
      // out = Vt [B, H, HD, S]; 64-col strip == one head (bn == h).
      const int b = tmrow / S;
      const int s = tmrow % S;
      v8h o;
#pragma unroll
      for (int r = 0; r < 8; ++r) o[r] = (_Float16)(acc[j][r] + bval);
      _Float16* dst = (_Float16*)out +
          (((size_t)b * kH + bn) * 64 + j * 16 + nn) * S + s + 8 * hi;
      *(v8h*)dst = o;
    } else {
#pragma unroll
      for (int r = 0; r < 8; ++r) {
        const int m = tmrow + r + hi * 8;
        float v = acc[j][r] + bval;
        if constexpr (MODE == EPI_F16_SCALE) v *= scale;
        if constexpr (MODE == EPI_F16_GELU)
          v = 0.5f * v * (1.0f + erff(v * 0.70710678118654752f));
        if constexpr (MODE == EPI_F32_RESID) {
          ((float*)out)[(size_t)m * N + n] = v + resid[(size_t)m * N + n];
        } else {
          ((_Float16*)out)[(size_t)m * N + n] = (_Float16)v;
        }
      }
    }
  }
}

// ---------------------------------------------------------------------------
// Flash attention, fully in registers (no LDS).
// scores^T = K_blk(A) x Q(B): query index in lanes, key index in VGPRs ->
// softmax is in-lane + one shfl_xor(16); exp'd P^T in C-layout is exactly the
// B-operand layout for ctx^T = V^T(A) x P^T(B).
// ---------------------------------------------------------------------------
__global__ __launch_bounds__(256) void attn_flash(
    const _Float16* __restrict__ q16,  // [B*T, D] (pre-scaled)
    const _Float16* __restrict__ k16,  // [B*S, D]
    const _Float16* __restrict__ vt16, // [B, H, HD, S]
    _Float16* __restrict__ ctx16) {    // [B*T, D]
  const int wave = blockIdx.x * (blockDim.x >> 5) + (threadIdx.x >> 5);
  const int ttiles = kT >> 4;
  const int tt = wave % ttiles;
  const int bh = wave / ttiles;
  const int h = bh % kH;
  const int b = bh / kH;
  const int lane = threadIdx.x & 31;
  const int hi = lane >> 4;
  const int nn = lane & 15;   // local query index

  const _Float16* qbase = q16 + ((size_t)(b * kT + tt * 16)) * kD + h * 64;
  const v16h qb0 = load_tile16x32(qbase, kD);
  const v16h qb1 = load_tile16x32(qbase + 32, kD);

  v8f O0 = {}, O1 = {}, O2 = {}, O3 = {};   // ctx^T: rows=hd, cols=q
  float mrun = -1e30f;
  float lrun = 0.0f;

  const _Float16* vbase = vt16 + ((size_t)(b * kH + h) * 64) * kS;

  for (int s0 = 0; s0 < kS; s0 += 32) {
    const _Float16* kb = k16 + ((size_t)(b * kS + s0)) * kD + h * 64;
    v8f c0 = {}, c1 = {};
    c0 = wmma_f16(load_tile16x32(kb, kD), qb0, c0);
    c0 = wmma_f16(load_tile16x32(kb + 32, kD), qb1, c0);
    c1 = wmma_f16(load_tile16x32(kb + (size_t)16 * kD, kD), qb0, c1);
    c1 = wmma_f16(load_tile16x32(kb + (size_t)16 * kD + 32, kD), qb1, c1);

    float bm = fmaxf(c0[0], c1[0]);
#pragma unroll
    for (int r = 1; r < 8; ++r) bm = fmaxf(bm, fmaxf(c0[r], c1[r]));
    bm = fmaxf(bm, __shfl_xor(bm, 16));
    const float nm = fmaxf(mrun, bm);
    const float sc = __expf(mrun - nm);
    mrun = nm;

    v16h pb;
    float bsum = 0.0f;
#pragma unroll
    for (int r = 0; r < 8; ++r) {
      const float p0 = __expf(c0[r] - nm);
      const float p1 = __expf(c1[r] - nm);
      bsum += p0 + p1;
      pb[r]     = (_Float16)p0;
      pb[r + 8] = (_Float16)p1;
    }
    lrun = lrun * sc + bsum;

#pragma unroll
    for (int r = 0; r < 8; ++r) { O0[r] *= sc; O1[r] *= sc; O2[r] *= sc; O3[r] *= sc; }

    const _Float16* vb = vbase + s0;
    O0 = wmma_f16(load_tile16x32(vb, kS), pb, O0);
    O1 = wmma_f16(load_tile16x32(vb + (size_t)16 * kS, kS), pb, O1);
    O2 = wmma_f16(load_tile16x32(vb + (size_t)32 * kS, kS), pb, O2);
    O3 = wmma_f16(load_tile16x32(vb + (size_t)48 * kS, kS), pb, O3);
  }

  lrun += __shfl_xor(lrun, 16);
  const float inv = 1.0f / lrun;
  _Float16* dst = ctx16 + ((size_t)(b * kT + tt * 16 + nn)) * kD + h * 64 + 8 * hi;
  v8h o0, o1, o2, o3;
#pragma unroll
  for (int r = 0; r < 8; ++r) {
    o0[r] = (_Float16)(O0[r] * inv);
    o1[r] = (_Float16)(O1[r] * inv);
    o2[r] = (_Float16)(O2[r] * inv);
    o3[r] = (_Float16)(O3[r] * inv);
  }
  *(v8h*)(dst)      = o0;
  *(v8h*)(dst + 16) = o1;
  *(v8h*)(dst + 32) = o2;
  *(v8h*)(dst + 48) = o3;
}

// ---------------------------------------------------------------------------
// LayerNorm over last dim (D=768). One block per row.
// ---------------------------------------------------------------------------
__global__ __launch_bounds__(256) void layernorm_row(
    const float* __restrict__ x, const float* __restrict__ gamma,
    const float* __restrict__ beta, float* __restrict__ outf,
    _Float16* __restrict__ outh, int D) {
  __shared__ float sh[64];
  const size_t row = blockIdx.x;
  const float* xr = x + row * D;
  float s = 0.0f, ss = 0.0f;
  for (int i = threadIdx.x; i < D; i += blockDim.x) {
    const float v = xr[i];
    s += v; ss += v * v;
  }
#pragma unroll
  for (int m = 16; m >= 1; m >>= 1) { s += __shfl_xor(s, m); ss += __shfl_xor(ss, m); }
  const int wid = threadIdx.x >> 5;
  if ((threadIdx.x & 31) == 0) { sh[wid] = s; sh[32 + wid] = ss; }
  __syncthreads();
  float ts = 0.0f, tss = 0.0f;
  const int nw = blockDim.x >> 5;
  for (int w = 0; w < nw; ++w) { ts += sh[w]; tss += sh[32 + w]; }
  const float mu = ts / D;
  const float var = tss / D - mu * mu;
  const float rstd = rsqrtf(var + 1e-5f);
  for (int i = threadIdx.x; i < D; i += blockDim.x) {
    const float v = (xr[i] - mu) * rstd * gamma[i] + beta[i];
    if (outh) outh[row * D + i] = (_Float16)v;
    else      outf[row * D + i] = v;
  }
}

// ---------------------------------------------------------------------------
// Host-side orchestration
// ---------------------------------------------------------------------------
extern "C" void kernel_launch(void* const* d_in, const int* in_sizes, int n_in,
                              void* d_out, int out_size, void* d_ws, size_t ws_size,
                              hipStream_t stream) {
  const float* query = (const float*)d_in[0];
  const float* key   = (const float*)d_in[1];
  const float* wq    = (const float*)d_in[2];
  const float* bq    = (const float*)d_in[3];
  const float* wk    = (const float*)d_in[4];
  const float* bk    = (const float*)d_in[5];
  const float* wv    = (const float*)d_in[6];
  const float* bv    = (const float*)d_in[7];
  const float* wo    = (const float*)d_in[8];
  const float* bo    = (const float*)d_in[9];
  const float* ln2_g = (const float*)d_in[10];
  const float* ln2_b = (const float*)d_in[11];
  const float* fc1_w = (const float*)d_in[12];
  const float* fc1_b = (const float*)d_in[13];
  const float* fc2_w = (const float*)d_in[14];
  const float* fc2_b = (const float*)d_in[15];
  const float* lnp_g = (const float*)d_in[16];
  const float* lnp_b = (const float*)d_in[17];

  const size_t nQ = (size_t)kB * kT * kD;   // 6.29M
  const size_t nK = (size_t)kB * kS * kD;   // 25.2M

  size_t off = 0;
  auto alloc = [&](size_t bytes) -> char* {
    char* p = (char*)d_ws + off;
    off += (bytes + 255) & ~(size_t)255;
    return p;
  };
  _Float16* wq16   = (_Float16*)alloc((size_t)kD * kD * 2);
  _Float16* wk16   = (_Float16*)alloc((size_t)kD * kD * 2);
  _Float16* wv16   = (_Float16*)alloc((size_t)kD * kD * 2);
  _Float16* wo16   = (_Float16*)alloc((size_t)kD * kD * 2);
  _Float16* fc1w16 = (_Float16*)alloc((size_t)kFF * kD * 2);
  _Float16* fc2w16 = (_Float16*)alloc((size_t)kD * kFF * 2);
  _Float16* qin16  = (_Float16*)alloc(nQ * 2);
  char*     kin_g  = alloc(nK * 2);           // key-f16, later reused as gelu act
  _Float16* kin16  = (_Float16*)kin_g;
  _Float16* g16    = (_Float16*)kin_g;        // alias (same size)
  _Float16* q16    = (_Float16*)alloc(nQ * 2);
  char*     k_y    = alloc(nK * 2);           // k-f16, later reused as fc2 out f32
  _Float16* k16    = (_Float16*)k_y;
  float*    ybuf   = (float*)k_y;             // alias (nQ*4 <= nK*2)
  _Float16* vt16   = (_Float16*)alloc(nK * 2);
  _Float16* ctx16  = (_Float16*)alloc(nQ * 2);
  float*    xbuf   = (float*)alloc(nQ * 4);
  _Float16* h16    = (_Float16*)alloc(nQ * 2);
  (void)ws_size; (void)in_sizes; (void)n_in; (void)out_size;

  auto cgrid = [](size_t n) { return dim3((unsigned)((n + 255) / 256)); };
  auto ggrid = [](int M, int N) { return dim3((unsigned)((size_t)(M / 128) * (N / 64))); };
  const dim3 blk(256);

  // 1) precision conversion (weights + activations)
  cvt_f32_f16<<<cgrid((size_t)kD * kD), blk, 0, stream>>>(wq, wq16, (size_t)kD * kD);
  cvt_f32_f16<<<cgrid((size_t)kD * kD), blk, 0, stream>>>(wk, wk16, (size_t)kD * kD);
  cvt_f32_f16<<<cgrid((size_t)kD * kD), blk, 0, stream>>>(wv, wv16, (size_t)kD * kD);
  cvt_f32_f16<<<cgrid((size_t)kD * kD), blk, 0, stream>>>(wo, wo16, (size_t)kD * kD);
  cvt_f32_f16<<<cgrid((size_t)kFF * kD), blk, 0, stream>>>(fc1_w, fc1w16, (size_t)kFF * kD);
  cvt_f32_f16<<<cgrid((size_t)kD * kFF), blk, 0, stream>>>(fc2_w, fc2w16, (size_t)kD * kFF);
  cvt_f32_f16<<<cgrid(nQ), blk, 0, stream>>>(query, qin16, nQ);
  cvt_f32_f16<<<cgrid(nK), blk, 0, stream>>>(key, kin16, nK);

  // 2) q/k/v projections (q pre-scaled by HD^-0.5, v stored transposed per head)
  gemm_wmma<EPI_F16_SCALE><<<ggrid(kB * kT, kD), blk, 0, stream>>>(
      qin16, wq16, bq, q16, nullptr, kB * kT, kD, kD, kScale, 0);
  gemm_wmma<EPI_F16><<<ggrid(kB * kS, kD), blk, 0, stream>>>(
      kin16, wk16, bk, k16, nullptr, kB * kS, kD, kD, 1.0f, 0);
  gemm_wmma<EPI_F16_TRANSV><<<ggrid(kB * kS, kD), blk, 0, stream>>>(
      kin16, wv16, bv, vt16, nullptr, kB * kS, kD, kD, 1.0f, kS);

  // 3) flash attention -> ctx
  {
    const int waves = kB * kH * (kT / 16);  // 6144, exact multiple of 8
    attn_flash<<<dim3(waves / 8), blk, 0, stream>>>(q16, k16, vt16, ctx16);
  }

  // 4) output projection + residual:  x = query + ctx @ wo^T + bo
  gemm_wmma<EPI_F32_RESID><<<ggrid(kB * kT, kD), blk, 0, stream>>>(
      ctx16, wo16, bo, xbuf, query, kB * kT, kD, kD, 1.0f, 0);

  // 5) LN2 -> h16
  layernorm_row<<<dim3(kB * kT), blk, 0, stream>>>(xbuf, ln2_g, ln2_b, nullptr, h16, kD);

  // 6) MLP: gelu(h @ fc1^T + b1) @ fc2^T + b2 + x
  gemm_wmma<EPI_F16_GELU><<<ggrid(kB * kT, kFF), blk, 0, stream>>>(
      h16, fc1w16, fc1_b, g16, nullptr, kB * kT, kFF, kD, 1.0f, 0);
  gemm_wmma<EPI_F32_RESID><<<ggrid(kB * kT, kD), blk, 0, stream>>>(
      g16, fc2w16, fc2_b, ybuf, xbuf, kB * kT, kD, kFF, 1.0f, 0);

  // 7) final LN -> f32 output
  layernorm_row<<<dim3(kB * kT), blk, 0, stream>>>(ybuf, lnp_g, lnp_b, (float*)d_out,
                                                   nullptr, kD);
}
